// AttnPool_73409581023420
// MI455X (gfx1250) — matrix operand order repaired and checked by reading, not compiled, not run on hardware
//
#include <hip/hip_runtime.h>
#include <hip/hip_bf16.h>
#include <math.h>

// Problem dims (fixed by reference): N=64, T=256, D=768, H=256; M = N*T.
#define Mtok 16384
#define Ddim 768
#define Hdim 256

typedef _Float16 v16h __attribute__((ext_vector_type(16)));
typedef _Float16 h8   __attribute__((ext_vector_type(8)));
typedef float    v8f  __attribute__((ext_vector_type(8)));
typedef unsigned int u32x4 __attribute__((ext_vector_type(4)));
typedef unsigned int u32x8 __attribute__((ext_vector_type(8)));

union H16 { v16h v; h8 h[2]; };

__device__ __forceinline__ v8f wmma16(v16h a, v16h b, v8f c) {
  // D = A(16x32 f16) * B(32x16 f16) + C(16x16 f32)
  return __builtin_amdgcn_wmma_f32_16x16x32_f16(false, a, false, b, (short)0, c,
                                                false, false);
}

// B-fragment: 16 contiguous halves (lane l16 = column n, elems e -> K = hi*16+e)
__device__ __forceinline__ v16h ldB16(const _Float16* p) {
  H16 u;
  u.h[0] = *(const h8*)p;
  u.h[1] = *(const h8*)(p + 8);
  return u.v;
}

// A-fragment from a row of 32 f16 (lane l16 = row m):
// elems e -> K = hi*16 + e
__device__ __forceinline__ v16h ldA(const _Float16* row, int hi) {
  H16 u;
  u.h[0] = *(const h8*)(row + hi * 16);
  u.h[1] = *(const h8*)(row + hi * 16 + 8);
  return u.v;
}

// A-fragment from f32 with on-the-fly f16 convert (same index map as ldA)
__device__ __forceinline__ v16h cvtA(const float* base, int hi) {
  v16h r;
  const float* p0 = base + hi * 16;
  const float* p1 = base + hi * 16 + 8;
#pragma unroll
  for (int i = 0; i < 8; ++i) {
    r[i]     = (_Float16)p0[i];
    r[i + 8] = (_Float16)p1[i];
  }
  return r;
}

// B-fragment from 16 contiguous f32 with convert
__device__ __forceinline__ v16h cvtB(const float* p) {
  v16h r;
#pragma unroll
  for (int i = 0; i < 16; ++i) r[i] = (_Float16)p[i];
  return r;
}

// ---------------------------------------------------------------------------
// Tensor Data Mover: async 2D tile load global -> LDS (CDNA5 TDM).
// Builds D# group0 (4 SGPRs) + group1 (8 SGPRs) and issues tensor_load_to_lds.
// data_size = 2 bytes (f16). Tracked by TENSORcnt.
// ---------------------------------------------------------------------------
__device__ __forceinline__ void tdm_load_2d_f16(
    unsigned lds_off, const void* gaddr, unsigned tensor_d0, unsigned tensor_d1,
    unsigned tile_d0, unsigned tile_d1, unsigned long long stride0) {
  unsigned long long ga = (unsigned long long)(uintptr_t)gaddr;
  u32x4 g0;
  g0[0] = 1u;                                     // count=1, no gather
  g0[1] = lds_off;                                // lds_addr (bytes)
  g0[2] = (unsigned)(ga & 0xffffffffu);           // global_addr[31:0]
  g0[3] = (unsigned)((ga >> 32) & 0x01ffffffu)    // global_addr[56:32]
          | (2u << 30);                           // type = 2 ("image")
  u32x8 g1;
  g1[0] = 0x00010000u;                            // data_size = 1 -> 2 bytes
  g1[1] = (tensor_d0 & 0xffffu) << 16;            // tensor_dim0[15:0]
  g1[2] = ((tensor_d0 >> 16) & 0xffffu) | ((tensor_d1 & 0xffffu) << 16);
  g1[3] = ((tensor_d1 >> 16) & 0xffffu) | ((tile_d0 & 0xffffu) << 16);
  g1[4] = tile_d1 & 0xffffu;                      // tile_dim1 (tile_dim2 = 0)
  g1[5] = (unsigned)(stride0 & 0xffffffffu);      // tensor_dim0_stride[31:0]
  g1[6] = (unsigned)((stride0 >> 32) & 0xffffu);  // stride[47:32]
  g1[7] = 0u;
  asm volatile("tensor_load_to_lds %0, %1" ::"s"(g0), "s"(g1) : "memory");
}

// ---------------------------------------------------------------------------
// Kernel 1: QKV projection.  One wave per 64x16 output tile (4 m-subtiles
// sharing each W B-fragment).  Q,K stored f16 row-major [M,H]; V stored
// transposed f16 [H,M].
// ---------------------------------------------------------------------------
__global__ __launch_bounds__(128) void qkv_proj(
    const float* __restrict__ x, const float* __restrict__ Wq,
    const float* __restrict__ Wk, const float* __restrict__ Wv,
    _Float16* __restrict__ Qm, _Float16* __restrict__ Km,
    _Float16* __restrict__ Vt) {
  const int wid  = blockIdx.x * 4 + (threadIdx.x >> 5);
  const int lane = threadIdx.x & 31;
  const int l16  = lane & 15;
  const int hi   = lane >> 4;
  const int m0   = (wid >> 4) * 64;   // 64-token-row tile
  const int h0   = (wid & 15) * 16;   // head-dim tile

  const float* xr[4];
#pragma unroll
  for (int j = 0; j < 4; ++j)
    xr[j] = x + (size_t)(m0 + j * 16 + l16) * Ddim;
  const float* qr = Wq + (size_t)(h0 + l16) * Ddim;
  const float* kr = Wk + (size_t)(h0 + l16) * Ddim;
  const float* vr = Wv + (size_t)(h0 + l16) * Ddim;

  v8f qa[4], ka[4], va[4];
#pragma unroll
  for (int j = 0; j < 4; ++j) {
    qa[j] = (v8f)0.0f; ka[j] = (v8f)0.0f; va[j] = (v8f)0.0f;
  }

  for (int d0 = 0; d0 < Ddim; d0 += 32) {
    v16h bq = cvtB(qr + d0 + hi * 16);
    v16h bk = cvtB(kr + d0 + hi * 16);
    v16h bv = cvtB(vr + d0 + hi * 16);
#pragma unroll
    for (int j = 0; j < 4; ++j) {
      v16h a = cvtA(xr[j] + d0, hi);
      qa[j] = wmma16(a, bq, qa[j]);
      ka[j] = wmma16(a, bk, ka[j]);
      va[j] = wmma16(a, bv, va[j]);
    }
  }

  // C-layout: elem r -> row m0 + j*16 + hi*8 + r, column h0 + l16
#pragma unroll
  for (int j = 0; j < 4; ++j) {
#pragma unroll
    for (int r = 0; r < 8; ++r) {
      const size_t m = (size_t)(m0 + j * 16 + hi * 8 + r);
      Qm[m * Hdim + h0 + l16] = (_Float16)qa[j][r];
      Km[m * Hdim + h0 + l16] = (_Float16)ka[j][r];
    }
    h8 vp;
#pragma unroll
    for (int r = 0; r < 8; ++r) vp[r] = (_Float16)va[j][r];
    *(h8*)(Vt + (size_t)(h0 + l16) * Mtok + m0 + j * 16 + hi * 8) = vp;
  }
}

// ---------------------------------------------------------------------------
// Kernel 2: flash attention + column-mean accumulation.
// 4 waves/block, 16 query rows each (Q fragments held in registers).
// K/V^T key-chunks (32 keys) staged into LDS by the Tensor Data Mover
// (wave 0), double-buffered so the DMA overlaps the 32 WMMAs per chunk.
// ---------------------------------------------------------------------------
__global__ __launch_bounds__(128) void flash_attn(
    const _Float16* __restrict__ Qm, const _Float16* __restrict__ Km,
    const _Float16* __restrict__ Vt, float* __restrict__ colsum) {
  __shared__ __align__(16) _Float16 Kbuf[2][32][Hdim];  // 2 x 16 KB
  __shared__ __align__(16) _Float16 Vbuf[2][Hdim][32];  // 2 x 16 KB
  __shared__ __align__(16) _Float16 plds[4][16][32];    // 4 KB

  const int wave = threadIdx.x >> 5;
  const int lane = threadIdx.x & 31;
  const int l16  = lane & 15;
  const int hi   = lane >> 4;
  const int qm0  = (blockIdx.x * 4 + wave) * 16;

  const _Float16* qrow = Qm + (size_t)(qm0 + l16) * Hdim;
  // Preload the wave's 16x256 Q tile as 8 A-fragments (loop invariant).
  v16h qf[8];
#pragma unroll
  for (int c = 0; c < 8; ++c) qf[c] = ldA(qrow + c * 32, hi);

  v8f accO[16];
#pragma unroll
  for (int t = 0; t < 16; ++t) accO[t] = (v8f)0.0f;
  float rowmax[8], rowsum[8];
#pragma unroll
  for (int r = 0; r < 8; ++r) { rowmax[r] = -1e30f; rowsum[r] = 0.0f; }

  const float scale = 0.0625f;  // 1/sqrt(H)
  const int nch = Mtok / 32;

  // Prologue: stage chunk 0 into buffer 0 (TDM, wave 0 only; EXEC-independent).
  if (threadIdx.x < 32) {
    tdm_load_2d_f16((unsigned)(uintptr_t)&Kbuf[0][0][0], Km, Hdim, Mtok, Hdim,
                    32, Hdim);
    tdm_load_2d_f16((unsigned)(uintptr_t)&Vbuf[0][0][0], Vt, Mtok, Hdim, 32,
                    Hdim, Mtok);
  }

  for (int i = 0; i < nch; ++i) {
    const int b = i & 1;
    // Issue next chunk's TDM pair, then wait for current pair (<=2 in flight).
    if (threadIdx.x < 32) {
      if (i + 1 < nch) {
        const int kn = (i + 1) * 32;
        tdm_load_2d_f16((unsigned)(uintptr_t)&Kbuf[b ^ 1][0][0],
                        Km + (size_t)kn * Hdim, Hdim, Mtok, Hdim, 32, Hdim);
        tdm_load_2d_f16((unsigned)(uintptr_t)&Vbuf[b ^ 1][0][0], Vt + kn, Mtok,
                        Hdim, 32, Hdim, Mtok);
        __builtin_amdgcn_s_wait_tensorcnt(2);
      } else {
        __builtin_amdgcn_s_wait_tensorcnt(0);
      }
    }
    __syncthreads();  // current buffer visible to all waves

    // ---- S = Q * K^T for 32 keys (two 16x16 tiles), K-dim = H in 8 chunks
    v8f Sa = (v8f)0.0f, Sb = (v8f)0.0f;
#pragma unroll
    for (int c = 0; c < 8; ++c) {
      Sa = wmma16(qf[c], ldB16(&Kbuf[b][l16][c * 32 + hi * 16]), Sa);
      Sb = wmma16(qf[c], ldB16(&Kbuf[b][16 + l16][c * 32 + hi * 16]), Sb);
    }

    // ---- online softmax (rows live in 16-lane groups of the C-layout)
    float cmax[8];
#pragma unroll
    for (int r = 0; r < 8; ++r) cmax[r] = fmaxf(Sa[r], Sb[r]) * scale;
#pragma unroll
    for (int m = 1; m <= 8; m <<= 1) {
#pragma unroll
      for (int r = 0; r < 8; ++r)
        cmax[r] = fmaxf(cmax[r], __shfl_xor(cmax[r], m, 32));
    }
    float alpha[8], psum[8], pa[8], pb[8];
#pragma unroll
    for (int r = 0; r < 8; ++r) {
      float nm = fmaxf(rowmax[r], cmax[r]);
      alpha[r] = __expf(rowmax[r] - nm);
      rowmax[r] = nm;
      pa[r] = __expf(Sa[r] * scale - nm);
      pb[r] = __expf(Sb[r] * scale - nm);
      psum[r] = pa[r] + pb[r];
    }
#pragma unroll
    for (int m = 1; m <= 8; m <<= 1) {
#pragma unroll
      for (int r = 0; r < 8; ++r) psum[r] += __shfl_xor(psum[r], m, 32);
    }
#pragma unroll
    for (int r = 0; r < 8; ++r) rowsum[r] = rowsum[r] * alpha[r] + psum[r];
#pragma unroll
    for (int t = 0; t < 16; ++t) {
#pragma unroll
      for (int r = 0; r < 8; ++r) accO[t][r] *= alpha[r];
    }

    // ---- transpose P (C-layout -> A-layout) through per-wave LDS tile
#pragma unroll
    for (int r = 0; r < 8; ++r) {
      plds[wave][hi * 8 + r][l16]      = (_Float16)pa[r];
      plds[wave][hi * 8 + r][16 + l16] = (_Float16)pb[r];
    }
    __syncthreads();
    v16h pf = ldA(&plds[wave][l16][0], hi);

    // ---- O += P * V  (16 tiles across H=256), V from LDS
#pragma unroll
    for (int t = 0; t < 16; ++t) {
      accO[t] = wmma16(pf, ldB16(&Vbuf[b][t * 16 + l16][hi * 16]), accO[t]);
    }
    __syncthreads();  // buffers free for reuse before next TDM issue
  }

  // ---- normalize rows and accumulate column sums of O (for the mean)
  float inv[8];
#pragma unroll
  for (int r = 0; r < 8; ++r) inv[r] = 1.0f / rowsum[r];
#pragma unroll
  for (int t = 0; t < 16; ++t) {
    float s = 0.0f;
#pragma unroll
    for (int r = 0; r < 8; ++r) s += accO[t][r] * inv[r];
    atomicAdd(&colsum[t * 16 + l16], s);
  }
}

// ---------------------------------------------------------------------------
// Kernel 0/3: init and final projection  out = (colsum/M) @ Wo^T
// ---------------------------------------------------------------------------
__global__ void zero_cols(float* colsum) { colsum[threadIdx.x] = 0.0f; }

__global__ void out_proj(const float* __restrict__ colsum,
                         const float* __restrict__ Wo,
                         float* __restrict__ out) {
  int d = blockIdx.x * blockDim.x + threadIdx.x;
  if (d < Ddim) {
    float s = 0.0f;
    for (int h = 0; h < Hdim; ++h) s += colsum[h] * Wo[(size_t)d * Hdim + h];
    out[d] = s * (1.0f / (float)Mtok);
  }
}

extern "C" void kernel_launch(void* const* d_in, const int* in_sizes, int n_in,
                              void* d_out, int out_size, void* d_ws,
                              size_t ws_size, hipStream_t stream) {
  const float* x  = (const float*)d_in[0];
  const float* Wq = (const float*)d_in[1];
  const float* Wk = (const float*)d_in[2];
  const float* Wv = (const float*)d_in[3];
  const float* Wo = (const float*)d_in[4];
  float* out = (float*)d_out;

  char* ws = (char*)d_ws;
  const size_t mat = (size_t)Mtok * Hdim * sizeof(_Float16);  // 8 MB each
  _Float16* Qm = (_Float16*)(ws);
  _Float16* Km = (_Float16*)(ws + mat);
  _Float16* Vt = (_Float16*)(ws + 2 * mat);
  float* colsum = (float*)(ws + 3 * mat);

  zero_cols<<<dim3(1), dim3(Hdim), 0, stream>>>(colsum);
  qkv_proj<<<dim3((Mtok / 64) * (Hdim / 16) / 4), dim3(128), 0, stream>>>(
      x, Wq, Wk, Wv, Qm, Km, Vt);
  flash_attn<<<dim3((Mtok / 16) / 4), dim3(128), 0, stream>>>(Qm, Km, Vt,
                                                              colsum);
  out_proj<<<dim3(3), dim3(256), 0, stream>>>(colsum, Wo, out);
}
